// conv_encoder_46265387712663
// MI455X (gfx1250) — compile-verified, matrix operand order checked by reading, not run on hardware
//
#include <hip/hip_runtime.h>

typedef __attribute__((ext_vector_type(16))) __bf16 v16bf;
typedef __attribute__((ext_vector_type(8)))  __bf16 v8bf;
typedef __attribute__((ext_vector_type(4)))  __bf16 v4bf;
typedef __attribute__((ext_vector_type(8)))  float  v8f;
typedef __attribute__((ext_vector_type(4)))  float  v4f;

namespace {
constexpr int kBatch  = 512;
constexpr int kNeur   = 512;
constexpr int kCh     = 384;
constexpr int kHW     = 169;            // 13*13
constexpr int kBM     = 64;             // batch tile
constexpr int kBN     = 128;            // neuron tile
constexpr int kSplit  = 16;             // K-split for parallelism (512 WGs total)
constexpr int kCPS    = kCh / kSplit;   // 24 channels per split
constexpr int kChunks = kCPS * 6;       // hw padded 169 -> 192 = 6 chunks of K=32
constexpr int kAst    = 40;             // A LDS row stride (bf16) — bank spread
constexpr int kBst    = 40;             // B LDS row stride (bf16)
constexpr int kWst    = 196;            // W_s LDS row stride (f32) — bank spread
}

// Exact f32 = hi(bf16, truncated top 16 bits) + lo(bf16 RNE of残り); residual ~2^-17.
__device__ __forceinline__ void splitf(float v, __bf16& h, __bf16& l) {
  unsigned u = __float_as_uint(v) & 0xffff0000u;
  h = __builtin_bit_cast(__bf16, (unsigned short)(u >> 16));
  l = (__bf16)(v - __uint_as_float(u));
}

// One WMMA 16-bit operand fragment: lane (m or n) = lane&15, K-half kb=(lane>>4)*8:
// elements [kb..kb+7] and [kb+16..kb+23] — two contiguous b128 LDS reads.
__device__ __forceinline__ v16bf ld_frag(const __bf16* p) {
  v8bf a = *(const v8bf*)p;
  v8bf b = *(const v8bf*)(p + 16);
  return __builtin_shufflevector(a, b, 0, 1, 2, 3, 4, 5, 6, 7,
                                 8, 9, 10, 11, 12, 13, 14, 15);
}

struct Stage {
  float xa[8];   // two A rows x 4 cols
  float d;       // W_d[n,c]
  float ws[16];  // W_s[n, hw0+j0 .. +15]
};

__global__ __launch_bounds__(256) void fused_factored_gemm(
    const float* __restrict__ xg,   // [512][384][169]
    const float* __restrict__ wsg,  // [512][169]
    const float* __restrict__ wdg,  // [512][384]
    float* __restrict__ part)       // [kSplit][512][512]
{
  __shared__ __align__(16) float  Wsl[kBN * kWst];      // 100352 B, zero-padded to 192 cols
  __shared__ __align__(16) __bf16 Ah[2][kBM * kAst];    // 10240 B
  __shared__ __align__(16) __bf16 Al[2][kBM * kAst];    // 10240 B
  __shared__ __align__(16) __bf16 Bh[2][kBN * kBst];    // 20480 B
  __shared__ __align__(16) __bf16 Bl[2][kBN * kBst];    // 20480 B   (total ~158 KB)

  const int tid = threadIdx.x;
  const int n0  = blockIdx.x * kBN;
  const int bm0 = blockIdx.y * kBM;
  const int ks  = blockIdx.z;
  const int c0  = ks * kCPS;

  // Cache W_s rows for this neuron tile, zero-padded hw 169..191 (kills all K-pad masking on B side).
  for (int idx = tid; idx < kBN * 192; idx += 256) {
    int n = idx / 192;
    int q = idx - n * 192;
    Wsl[n * kWst + q] = (q < kHW) ? wsg[(size_t)(n0 + n) * kHW + q] : 0.0f;
  }
  __syncthreads();

  const int g    = tid & 7;         // A col group (4 floats)
  const int mrow = tid >> 3;        // A row 0..31 (also handles row+32)
  const int bn   = tid >> 1;        // B neuron row 0..127
  const int j0   = (tid & 1) * 16;  // B col base

  Stage rg;

  auto stage_load = [&](int i) {
    const int c   = c0 + i / 6;
    const int hw0 = (i % 6) * 32;
    const size_t rb0 = ((size_t)(bm0 + mrow) * kCh + c) * kHW + hw0 + g * 4;
    const size_t rb1 = rb0 + (size_t)32 * kCh * kHW;
#pragma unroll
    for (int e = 0; e < 4; ++e) {
      const bool ok = (hw0 + g * 4 + e) < kHW;  // mask K-pad on A side
      rg.xa[e]     = ok ? xg[rb0 + e] : 0.0f;
      rg.xa[4 + e] = ok ? xg[rb1 + e] : 0.0f;
    }
    rg.d = wdg[(size_t)(n0 + bn) * kCh + c];
#pragma unroll
    for (int q = 0; q < 4; ++q) {
      v4f s = *(const v4f*)&Wsl[bn * kWst + hw0 + j0 + 4 * q];
#pragma unroll
      for (int e = 0; e < 4; ++e) rg.ws[4 * q + e] = s[e];
    }
  };

  auto stage_store = [&](int buf) {
    v4bf h0, l0, h1, l1;
#pragma unroll
    for (int e = 0; e < 4; ++e) {
      __bf16 h, l;
      splitf(rg.xa[e], h, l);       h0[e] = h; l0[e] = l;
      splitf(rg.xa[4 + e], h, l);   h1[e] = h; l1[e] = l;
    }
    *(v4bf*)&Ah[buf][mrow * kAst + g * 4] = h0;
    *(v4bf*)&Al[buf][mrow * kAst + g * 4] = l0;
    *(v4bf*)&Ah[buf][(mrow + 32) * kAst + g * 4] = h1;
    *(v4bf*)&Al[buf][(mrow + 32) * kAst + g * 4] = l1;

    v8bf bh0, bh1, bl0, bl1;
#pragma unroll
    for (int q = 0; q < 16; ++q) {
      __bf16 h, l;
      splitf(rg.d * rg.ws[q], h, l);  // B_eff[n,k] = W_d[n,c]*W_s[n,hw], split in f32
      if (q < 8) { bh0[q] = h; bl0[q] = l; }
      else       { bh1[q - 8] = h; bl1[q - 8] = l; }
    }
    *(v8bf*)&Bh[buf][bn * kBst + j0]     = bh0;
    *(v8bf*)&Bh[buf][bn * kBst + j0 + 8] = bh1;
    *(v8bf*)&Bl[buf][bn * kBst + j0]     = bl0;
    *(v8bf*)&Bl[buf][bn * kBst + j0 + 8] = bl1;
  };

  // 8 waves, each owns a 32x32 output macro-tile (2x2 WMMA sub-tiles) for frag reuse.
  const int lane = tid & 31;
  const int wv   = tid >> 5;
  const int wm   = wv >> 2;             // 0..1
  const int wn   = wv & 3;              // 0..3
  const int r16  = lane & 15;
  const int kb   = (lane >> 4) * 8;

  v8f acc[2][2] = {};

  stage_load(0);
  stage_store(0);

  for (int i = 0; i < kChunks; ++i) {
    __syncthreads();
    const int  buf  = i & 1;
    const bool more = (i + 1) < kChunks;
    if (more) stage_load(i + 1);        // global loads in flight behind the WMMAs

    v16bf afh[2], afl[2], bfh[2], bfl[2];
#pragma unroll
    for (int sm = 0; sm < 2; ++sm) {
      const int row = (wm * 32 + sm * 16 + r16) * kAst + kb;
      afh[sm] = ld_frag(&Ah[buf][row]);
      afl[sm] = ld_frag(&Al[buf][row]);
    }
#pragma unroll
    for (int sn = 0; sn < 2; ++sn) {
      const int row = (wn * 32 + sn * 16 + r16) * kBst + kb;
      bfh[sn] = ld_frag(&Bh[buf][row]);
      bfl[sn] = ld_frag(&Bl[buf][row]);
    }
#pragma unroll
    for (int sm = 0; sm < 2; ++sm)
#pragma unroll
      for (int sn = 0; sn < 2; ++sn) {
        v8f a = acc[sm][sn];
        a = __builtin_amdgcn_wmma_f32_16x16x32_bf16(false, afh[sm], false, bfh[sn], (short)0, a, false, false);
        a = __builtin_amdgcn_wmma_f32_16x16x32_bf16(false, afh[sm], false, bfl[sn], (short)0, a, false, false);
        a = __builtin_amdgcn_wmma_f32_16x16x32_bf16(false, afl[sm], false, bfh[sn], (short)0, a, false, false);
        acc[sm][sn] = a;
      }

    if (more) stage_store((i + 1) & 1); // convert+write next chunk into other buffer
  }

  // Epilogue: C layout — VGPR r holds (M = r + 8*(lane>>4), N = lane&15).
  const int rowoff = (lane >> 4) * 8;
#pragma unroll
  for (int sm = 0; sm < 2; ++sm)
#pragma unroll
    for (int sn = 0; sn < 2; ++sn) {
      const int row0 = bm0 + wm * 32 + sm * 16 + rowoff;
      const int col  = n0 + wn * 32 + sn * 16 + r16;
      float* pp = part + ((size_t)ks * kBatch + row0) * kNeur + col;
#pragma unroll
      for (int r = 0; r < 8; ++r) pp[(size_t)r * kNeur] = acc[sm][sn][r];
    }
}

__global__ __launch_bounds__(256) void reduce_bias_k(
    const float* __restrict__ part, const float* __restrict__ wb,
    float* __restrict__ out)
{
  const int i = blockIdx.x * 256 + threadIdx.x;  // b*512 + n
  float s = wb[i & (kNeur - 1)];
#pragma unroll
  for (int ks = 0; ks < kSplit; ++ks)
    s += part[(size_t)ks * (kBatch * kNeur) + i];
  out[i] = s;
}

extern "C" void kernel_launch(void* const* d_in, const int* in_sizes, int n_in,
                              void* d_out, int out_size, void* d_ws, size_t ws_size,
                              hipStream_t stream) {
  (void)in_sizes; (void)n_in; (void)out_size; (void)ws_size;
  const float* x  = (const float*)d_in[0];  // [512][384][13][13]
  const float* Ws = (const float*)d_in[1];  // [512][13][13]
  const float* Wd = (const float*)d_in[2];  // [512][384]
  const float* Wb = (const float*)d_in[3];  // [1][512]
  float* out  = (float*)d_out;              // [512][512]
  float* part = (float*)d_ws;               // kSplit * 512 * 512 * 4 = 16 MB scratch

  dim3 grid(kNeur / kBN, kBatch / kBM, kSplit);  // (4, 8, 16) = 512 workgroups
  fused_factored_gemm<<<grid, 256, 0, stream>>>(x, Ws, Wd, part);
  reduce_bias_k<<<dim3((kBatch * kNeur) / 256), 256, 0, stream>>>(part, Wb, out);
}
